// MH_Lori_19490561589718
// MI455X (gfx1250) — matrix operand order repaired and checked by reading, not compiled, not run on hardware
//
#include <hip/hip_runtime.h>
#include <hip/hip_bf16.h>

// ---------------------------------------------------------------------------
// Problem constants (from reference)
// ---------------------------------------------------------------------------
#define B_      4
#define S_      4096
#define HID_    1024
#define HEADS_  8
#define D_      128
#define NSEG_   16
#define SEGLEN_ 256
#define E_      16
#define I_      512
#define NGRP_   (B_ * NSEG_ * HEADS_)   // 512 (b,n,h) groups

typedef __attribute__((ext_vector_type(8)))  float   v8f;
typedef __attribute__((ext_vector_type(8)))  __bf16  v8bf;
typedef __attribute__((ext_vector_type(16))) __bf16  v16bf;

#define PADK 40   // LDS row stride (bf16) for 32-wide K tiles; spreads banks

__device__ __forceinline__ v8f wmma_bf16(v16bf a, v16bf b, v8f c) {
    return __builtin_amdgcn_wmma_f32_16x16x32_bf16(false, a, false, b,
                                                   (short)0, c, false, false);
}

__device__ __forceinline__ v16bf cat8(v8bf a, v8bf b) {
    return __builtin_shufflevector(a, b, 0, 1, 2, 3, 4, 5, 6, 7,
                                         8, 9, 10, 11, 12, 13, 14, 15);
}

// ---------------------------------------------------------------------------
// Split-bf16 batched WMMA GEMM (3-product error-corrected):
//   C[M,N] = act( (Ahi+Alo)[M,K] @ (Whi+Wlo)[N,K]^T + bias )
// computed as  hi*hi + lo*hi + hi*lo  accumulated in fp32 WMMA accumulators.
// Assumes M%64==0, N%128==0, K%32==0 (true for all call sites),
// lda=ldw=K, ldc=N. Block=256 threads (8 wave32); block tile 64x128;
// wave tile 32x32 (2x2 of 16x16); K staged through LDS 32 wide.
//
// ISA bf16 fragment layout (CDNA5 ISA):
//   A 16x32: lane%16 = M row; lane<16 holds K [0..7]+[16..23], lane>=16
//            holds K [8..15]+[24..31]  -> two 16B LDS reads per fragment.
//   B 32x16: lane%16 = N col; lane<16 holds K [0..7]+[16..23], lane>=16
//            holds K [8..15]+[24..31] (same as A; two 16B LDS reads).
//   C 16x16: VGPR r -> M = r + (lane<16?0:8), N = lane%16
// ---------------------------------------------------------------------------
template<bool RELU, bool BIAS, bool SPLIT_OUT>
__global__ __launch_bounds__(256)
void wmma_gemm_bf16x3(const __bf16* __restrict__ Ahi, const __bf16* __restrict__ Alo,
                      const __bf16* __restrict__ Whi, const __bf16* __restrict__ Wlo,
                      const float*  __restrict__ bias,
                      float*  __restrict__ C,
                      __bf16* __restrict__ Chi, __bf16* __restrict__ Clo,
                      int M, int N, int K,
                      long long strideA, long long strideW, long long strideC)
{
    const long long batch = blockIdx.z;
    Ahi += batch * strideA;  Alo += batch * strideA;
    Whi += batch * strideW;  Wlo += batch * strideW;
    if (SPLIT_OUT) { Chi += batch * strideC; Clo += batch * strideC; }
    else           { C   += batch * strideC; }

    __shared__ __bf16 sAh[64 * PADK];    // 5.0 KB per plane
    __shared__ __bf16 sAl[64 * PADK];
    __shared__ __bf16 sWh[128 * PADK];   // 10.0 KB per plane
    __shared__ __bf16 sWl[128 * PADK];

    const int tid  = threadIdx.x;     // 0..255
    const int lane = tid & 31;
    const int wave = tid >> 5;        // 0..7
    const int wm   = wave & 1;        // 2 waves along M (32 rows each)
    const int wn   = wave >> 1;       // 4 waves along N (32 cols each)
    const int bm   = blockIdx.y * 64;
    const int bn   = blockIdx.x * 128;

    v8f acc00 = {}, acc01 = {}, acc10 = {}, acc11 = {};

    // Tile loaders: one 16B chunk of A per plane, two 16B chunks of W per plane.
    const int arow = tid >> 2;          // 0..63
    const int acol = (tid & 3) * 8;     // 0,8,16,24 (bf16 columns)

    // Fragment addressing
    const int fr = lane & 15;           // A: M row / B: N (= W row)
    const int kA = (lane >> 4) * 8;     // A chunk base (second chunk at +16)
    const int kB = (lane >> 4) * 8;     // B chunk base (second chunk at +16)

    for (int k0 = 0; k0 < K; k0 += 32) {
        v8bf ah = *(const v8bf*)(Ahi + (long long)(bm + arow) * K + k0 + acol);
        v8bf al = *(const v8bf*)(Alo + (long long)(bm + arow) * K + k0 + acol);
        v8bf wh0 = *(const v8bf*)(Whi + (long long)(bn + arow) * K + k0 + acol);
        v8bf wh1 = *(const v8bf*)(Whi + (long long)(bn + 64 + arow) * K + k0 + acol);
        v8bf wl0 = *(const v8bf*)(Wlo + (long long)(bn + arow) * K + k0 + acol);
        v8bf wl1 = *(const v8bf*)(Wlo + (long long)(bn + 64 + arow) * K + k0 + acol);

        __syncthreads();   // previous iteration's fragment reads complete
        *(v8bf*)(sAh + arow * PADK + acol)        = ah;
        *(v8bf*)(sAl + arow * PADK + acol)        = al;
        *(v8bf*)(sWh + arow * PADK + acol)        = wh0;
        *(v8bf*)(sWh + (64 + arow) * PADK + acol) = wh1;
        *(v8bf*)(sWl + arow * PADK + acol)        = wl0;
        *(v8bf*)(sWl + (64 + arow) * PADK + acol) = wl1;
        __syncthreads();

        // ---- load all fragments for this 32-deep slab ----
        const int r0 = (wm * 32 + fr) * PADK;
        const int r1 = (wm * 32 + 16 + fr) * PADK;
        v16bf ah0 = cat8(*(const v8bf*)(sAh + r0 + kA), *(const v8bf*)(sAh + r0 + kA + 16));
        v16bf al0 = cat8(*(const v8bf*)(sAl + r0 + kA), *(const v8bf*)(sAl + r0 + kA + 16));
        v16bf ah1 = cat8(*(const v8bf*)(sAh + r1 + kA), *(const v8bf*)(sAh + r1 + kA + 16));
        v16bf al1 = cat8(*(const v8bf*)(sAl + r1 + kA), *(const v8bf*)(sAl + r1 + kA + 16));

        const int w0 = (wn * 32 + fr) * PADK;
        const int w1 = (wn * 32 + 16 + fr) * PADK;
        v16bf bh0 = cat8(*(const v8bf*)(sWh + w0 + kB), *(const v8bf*)(sWh + w0 + kB + 16));
        v16bf bl0 = cat8(*(const v8bf*)(sWl + w0 + kB), *(const v8bf*)(sWl + w0 + kB + 16));
        v16bf bh1 = cat8(*(const v8bf*)(sWh + w1 + kB), *(const v8bf*)(sWh + w1 + kB + 16));
        v16bf bl1 = cat8(*(const v8bf*)(sWl + w1 + kB), *(const v8bf*)(sWl + w1 + kB + 16));

        // ---- 3-product accumulation: hi*hi + lo*hi + hi*lo ----
        acc00 = wmma_bf16(ah0, bh0, acc00);
        acc00 = wmma_bf16(al0, bh0, acc00);
        acc00 = wmma_bf16(ah0, bl0, acc00);

        acc01 = wmma_bf16(ah0, bh1, acc01);
        acc01 = wmma_bf16(al0, bh1, acc01);
        acc01 = wmma_bf16(ah0, bl1, acc01);

        acc10 = wmma_bf16(ah1, bh0, acc10);
        acc10 = wmma_bf16(al1, bh0, acc10);
        acc10 = wmma_bf16(ah1, bl0, acc10);

        acc11 = wmma_bf16(ah1, bh1, acc11);
        acc11 = wmma_bf16(al1, bh1, acc11);
        acc11 = wmma_bf16(ah1, bl1, acc11);
    }

    // Epilogue
    const int mo = (lane >> 4) * 8;
    #pragma unroll
    for (int mi = 0; mi < 2; ++mi) {
        #pragma unroll
        for (int ni = 0; ni < 2; ++ni) {
            const v8f& acc = (mi == 0) ? (ni == 0 ? acc00 : acc01)
                                       : (ni == 0 ? acc10 : acc11);
            const int col = bn + wn * 32 + ni * 16 + fr;
            float bv = 0.0f;
            if (BIAS) bv = bias[col];
            #pragma unroll
            for (int r = 0; r < 8; ++r) {
                const int row = bm + wm * 32 + mi * 16 + mo + r;
                float v = acc[r] + bv;
                if (RELU) v = fmaxf(v, 0.0f);
                const long long idx = (long long)row * N + col;
                if (SPLIT_OUT) {
                    __bf16 h = (__bf16)v;
                    Chi[idx] = h;
                    Clo[idx] = (__bf16)(v - (float)h);
                } else {
                    C[idx] = v;
                }
            }
        }
    }
}

// ---------------------------------------------------------------------------
// f32 -> (bf16 hi, bf16 lo) split of external inputs.
// ---------------------------------------------------------------------------
__global__ __launch_bounds__(256)
void split_f32_kernel(const float* __restrict__ src,
                      __bf16* __restrict__ hi, __bf16* __restrict__ lo)
{
    const long long idx = (long long)blockIdx.x * 256 + threadIdx.x;
    float v = src[idx];
    __bf16 h = (__bf16)v;
    hi[idx] = h;
    lo[idx] = (__bf16)(v - (float)h);
}

// ---------------------------------------------------------------------------
// Segment mean + routing softmax (reads y = yhi + ylo).
//   y viewed (B,NSEG,SEGLEN,HEADS,D); mean over SEGLEN; logits = avg @ emb(D,E);
//   ew = softmax_e. One block per g=(b*NSEG+n)*HEADS+h, 128 threads.
// ---------------------------------------------------------------------------
__global__ __launch_bounds__(128)
void avg_softmax_kernel(const __bf16* __restrict__ yhi,
                        const __bf16* __restrict__ ylo,
                        const float* __restrict__ emb,   // (D,E)
                        float* __restrict__ ew)          // (NGRP,E)
{
    const int g  = blockIdx.x;
    const int h  = g % HEADS_;
    const int bn = g / HEADS_;                 // b*NSEG + n

    __shared__ float avg_s[D_];
    __shared__ float logit_s[E_];

    const int d = threadIdx.x;                 // 0..127
    const long long base = ((long long)bn * SEGLEN_ * HEADS_ + h) * D_ + d;
    float s = 0.0f;
    #pragma unroll 4
    for (int l = 0; l < SEGLEN_; ++l) {
        const long long off = base + (long long)l * (HEADS_ * D_);
        s += (float)yhi[off] + (float)ylo[off];
    }
    avg_s[d] = s * (1.0f / SEGLEN_);
    __syncthreads();

    if (d < E_) {
        float lg = 0.0f;
        #pragma unroll 8
        for (int dd = 0; dd < D_; ++dd)
            lg += avg_s[dd] * emb[dd * E_ + d];
        logit_s[d] = lg;
    }
    __syncthreads();

    if (d < E_) {
        float mx = logit_s[0];
        #pragma unroll
        for (int e = 1; e < E_; ++e) mx = fmaxf(mx, logit_s[e]);
        float sum = 0.0f;
        #pragma unroll
        for (int e = 0; e < E_; ++e) sum += expf(logit_s[e] - mx);
        ew[g * E_ + d] = expf(logit_s[d] - mx) / sum;
    }
}

// ---------------------------------------------------------------------------
// Expert mixing with causal segment shift folded in, emitting split bf16:
//   m[g, idx] = sum_e Wsrc[e, idx] * ew[gsrc, e],  gsrc = (b, max(n-1,0), h)
// ---------------------------------------------------------------------------
__global__ __launch_bounds__(256)
void mix_kernel(const float* __restrict__ Wsrc,   // (E, RC)
                const float* __restrict__ ew,     // (NGRP, E)
                __bf16* __restrict__ outhi,       // (NGRP, RC)
                __bf16* __restrict__ outlo,
                int RC)
{
    const int g = blockIdx.y;
    const int h = g % HEADS_;
    const int n = (g / HEADS_) % NSEG_;
    const int b = g / (HEADS_ * NSEG_);
    const int ns = (n == 0) ? 0 : (n - 1);
    const int gs = (b * NSEG_ + ns) * HEADS_ + h;

    __shared__ float ws[E_];
    if (threadIdx.x < E_) ws[threadIdx.x] = ew[gs * E_ + threadIdx.x];
    __syncthreads();

    const long long idx = (long long)blockIdx.x * 256 + threadIdx.x;
    float acc = 0.0f;
    #pragma unroll
    for (int e = 0; e < E_; ++e)
        acc = fmaf(Wsrc[(long long)e * RC + idx], ws[e], acc);

    const long long o = (long long)g * RC + idx;
    __bf16 hv = (__bf16)acc;
    outhi[o] = hv;
    outlo[o] = (__bf16)(acc - (float)hv);
}

// ---------------------------------------------------------------------------
// Orchestration. Workspace (bf16 hi/lo pairs unless noted):
//   x 2x16.7M | Wmh 2x1M | Wmg 2x1M | y 2x16.7M | m1u 2x33.5M | m2u 2x33.5M
//   h 2x67.1M | z 2x16.7M | ew 8K f32   => ~746 MB of d_ws.
// ---------------------------------------------------------------------------
extern "C" void kernel_launch(void* const* d_in, const int* in_sizes, int n_in,
                              void* d_out, int out_size, void* d_ws, size_t ws_size,
                              hipStream_t stream)
{
    const float* x       = (const float*)d_in[0];  // (B,S,HID)
    const float* W_mh    = (const float*)d_in[1];  // (HID,HID)
    const float* b_mh    = (const float*)d_in[2];  // (HID)
    const float* W_merge = (const float*)d_in[3];  // (HID,HID)
    const float* b_merge = (const float*)d_in[4];  // (HID)
    const float* emb     = (const float*)d_in[5];  // (D,E)
    const float* flin    = (const float*)d_in[6];  // (E,I,D)
    const float* slin    = (const float*)d_in[7];  // (E,D,I)
    float* out = (float*)d_out;
    (void)in_sizes; (void)n_in; (void)out_size; (void)ws_size;

    const long long NY   = (long long)B_ * S_ * HID_;        // 16,777,216
    const long long NW   = (long long)HID_ * HID_;           //  1,048,576
    const long long NMIX = (long long)NGRP_ * I_ * D_;       // 33,554,432
    const long long NH   = (long long)NGRP_ * SEGLEN_ * I_;  // 67,108,864

    char* p = (char*)d_ws;
    auto alloc = [&](long long bytes) -> char* {
        char* r = p; p += (bytes + 255) & ~255LL; return r;
    };
    __bf16* xhi  = (__bf16*)alloc(NY * 2);
    __bf16* xlo  = (__bf16*)alloc(NY * 2);
    __bf16* wmhh = (__bf16*)alloc(NW * 2);
    __bf16* wmhl = (__bf16*)alloc(NW * 2);
    __bf16* wmgh = (__bf16*)alloc(NW * 2);
    __bf16* wmgl = (__bf16*)alloc(NW * 2);
    __bf16* yhi  = (__bf16*)alloc(NY * 2);
    __bf16* ylo  = (__bf16*)alloc(NY * 2);
    __bf16* m1hi = (__bf16*)alloc(NMIX * 2);
    __bf16* m1lo = (__bf16*)alloc(NMIX * 2);
    __bf16* m2hi = (__bf16*)alloc(NMIX * 2);
    __bf16* m2lo = (__bf16*)alloc(NMIX * 2);
    __bf16* hhi  = (__bf16*)alloc(NH * 2);
    __bf16* hlo  = (__bf16*)alloc(NH * 2);
    __bf16* zhi  = (__bf16*)alloc(NY * 2);
    __bf16* zlo  = (__bf16*)alloc(NY * 2);
    float*  ew   = (float*)alloc((long long)NGRP_ * E_ * 4);

    // 0) split external f32 inputs into bf16 hi/lo planes
    split_f32_kernel<<<(unsigned)(NY / 256), 256, 0, stream>>>(x, xhi, xlo);
    split_f32_kernel<<<(unsigned)(NW / 256), 256, 0, stream>>>(W_mh, wmhh, wmhl);
    split_f32_kernel<<<(unsigned)(NW / 256), 256, 0, stream>>>(W_merge, wmgh, wmgl);

    // 1) y = x @ W_mh^T + b_mh   (M=16384, N=1024, K=1024) -> split output
    wmma_gemm_bf16x3<false, true, true>
        <<<dim3(HID_ / 128, (B_ * S_) / 64, 1), 256, 0, stream>>>(
        xhi, xlo, wmhh, wmhl, b_mh, nullptr, yhi, ylo,
        B_ * S_, HID_, HID_, 0, 0, 0);

    // 2) routing weights ew
    avg_softmax_kernel<<<NGRP_, 128, 0, stream>>>(yhi, ylo, emb, ew);

    // 3) mixed (and causally shifted) per-group expert weights, split output
    mix_kernel<<<dim3((I_ * D_) / 256, NGRP_), 256, 0, stream>>>(flin, ew, m1hi, m1lo, I_ * D_);
    mix_kernel<<<dim3((I_ * D_) / 256, NGRP_), 256, 0, stream>>>(slin, ew, m2hi, m2lo, I_ * D_);

    // 4) h = relu(yr @ m1u^T) per group (M=256, N=512, K=128)
    //    yr = y viewed (B,NSEG,HEADS,SEGLEN,D): contiguous 256x128 per group.
    wmma_gemm_bf16x3<true, false, true>
        <<<dim3(I_ / 128, SEGLEN_ / 64, NGRP_), 256, 0, stream>>>(
        yhi, ylo, m1hi, m1lo, nullptr, nullptr, hhi, hlo,
        SEGLEN_, I_, D_,
        (long long)SEGLEN_ * D_, (long long)I_ * D_, (long long)SEGLEN_ * I_);

    // 5) z = h @ m2u^T per group (M=256, N=128, K=512)
    wmma_gemm_bf16x3<false, false, true>
        <<<dim3(D_ / 128, SEGLEN_ / 64, NGRP_), 256, 0, stream>>>(
        hhi, hlo, m2hi, m2lo, nullptr, nullptr, zhi, zlo,
        SEGLEN_, D_, I_,
        (long long)SEGLEN_ * I_, (long long)I_ * D_, (long long)SEGLEN_ * D_);

    // 6) out = z @ W_merge^T + b_merge  (flat reshape of z is identity), f32 out
    wmma_gemm_bf16x3<false, true, false>
        <<<dim3(HID_ / 128, (B_ * S_) / 64, 1), 256, 0, stream>>>(
        zhi, zlo, wmgh, wmgl, b_merge, out, nullptr, nullptr,
        B_ * S_, HID_, HID_, 0, 0, 0);
}
